// ScaledDotProductAttention_28020366639692
// MI455X (gfx1250) — compile-verified, matrix operand order checked by reading, not run on hardware
//
#include <hip/hip_runtime.h>

// ---------------------------------------------------------------------------
// Flash-attention forward (causal) for gfx1250 / CDNA5, bf16 WMMA fp32-accum.
// B=4 H=16 S=2048 D=128.
// V is staged TRANSPOSED in LDS so every WMMA B-fragment (K and V) is two
// contiguous 16-byte ds loads -> no scalar u16 gather / packing moves.
// ---------------------------------------------------------------------------

typedef __attribute__((ext_vector_type(16))) __bf16 v16bf;
typedef __attribute__((ext_vector_type(8)))  __bf16 v8bf;
typedef __attribute__((ext_vector_type(8)))  float  v8f;
typedef __attribute__((ext_vector_type(8)))  float  v8fl;
typedef __attribute__((ext_vector_type(2)))  float  v2f;

constexpr int Bn = 4;
constexpr int Hn = 16;
constexpr int Sn = 2048;
constexpr int Dn = 128;

constexpr int BM = 64;        // q rows per workgroup (4 waves x 16)
constexpr int BN = 32;        // k positions per inner step
constexpr int KROW = Dn + 8;  // LDS pitch (bf16) for K tile rows [k][d]
constexpr int VROW = BN + 8;  // LDS pitch (bf16) for V^T tile rows [d][k]
constexpr int PROW = BN + 8;  // LDS pitch (bf16) for P scratch

__device__ __forceinline__ v8f zero8() {
  v8f z;
#pragma unroll
  for (int i = 0; i < 8; ++i) z[i] = 0.0f;
  return z;
}

__device__ __forceinline__ v8f wmma_bf16(v16bf a, v16bf b, v8f c) {
  // D = A(16x32 bf16) * B(32x16 bf16) + C(16x16 f32)
  return __builtin_amdgcn_wmma_f32_16x16x32_bf16(
      /*neg_a=*/false, a, /*neg_b=*/false, b,
      /*c_mod=*/(short)0, c, /*reuse_a=*/false, /*reuse_b=*/false);
}

// Read a 16-element bf16 A/B fragment from one LDS row.
// Lane group kb=0 holds K = [0..7] U [16..23]; kb=8 holds [8..15] U [24..31].
__device__ __forceinline__ v16bf frag16(const __bf16* row, int kb) {
  v8bf lo = *(const v8bf*)(row + kb);
  v8bf hi = *(const v8bf*)(row + 16 + kb);
  v16bf r;
#pragma unroll
  for (int e = 0; e < 8; ++e) { r[e] = lo[e]; r[e + 8] = hi[e]; }
  return r;
}

// Cooperative (128-thread) staging of one BN x D fp32 tile:
//   K -> LDS bf16 row-major  [k][d]   (pitch KROW)
//   V -> LDS bf16 TRANSPOSED [d][k]   (pitch VROW)
__device__ __forceinline__ void stage_tile(__bf16* kd, __bf16* vtd,
                                           const float* __restrict__ Kb,
                                           const float* __restrict__ Vb,
                                           int kpos, int tid) {
  // ---- K: 4 threads per row, 32 contiguous columns each -------------------
  {
    const int r  = tid >> 2;        // 0..31
    const int cb = (tid & 3) * 32;  // 0,32,64,96
    const float* kg = Kb + (size_t)(kpos + r) * Dn + cb;
    __bf16* kdr = kd + r * KROW + cb;
    if (kpos + BN + 31 < Sn) {
      __builtin_prefetch(kg + BN * Dn, 0, 0);  // global_prefetch_b8
    }
#pragma unroll
    for (int j = 0; j < 4; ++j) {
      v8fl kf = *(const v8fl*)(kg + j * 8);
      v8bf kh;
#pragma unroll
      for (int e = 0; e < 8; ++e) kh[e] = (__bf16)kf[e];
      *(v8bf*)(kdr + j * 8) = kh;
    }
  }
  // ---- V transposed: thread owns d-pair (t&63) and k-group (t>>6) ---------
  {
    const int d0 = (tid & 63) * 2;  // 0,2,..,126
    const int kg = tid >> 6;        // 0..1 -> k rows kg*16..kg*16+15
    const float* vg = Vb + (size_t)(kpos + kg * 16) * Dn + d0;
    if (kpos + BN + 31 < Sn) {
      __builtin_prefetch(vg + BN * Dn, 0, 0);
    }
    v8bf a0lo, a0hi, a1lo, a1hi;
#pragma unroll
    for (int k = 0; k < 8; ++k) {
      v2f f = *(const v2f*)(vg + (size_t)k * Dn);
      a0lo[k] = (__bf16)f.x;
      a1lo[k] = (__bf16)f.y;
    }
#pragma unroll
    for (int k = 0; k < 8; ++k) {
      v2f f = *(const v2f*)(vg + (size_t)(k + 8) * Dn);
      a0hi[k] = (__bf16)f.x;
      a1hi[k] = (__bf16)f.y;
    }
    __bf16* v0 = vtd + d0 * VROW + kg * 16;
    __bf16* v1 = vtd + (d0 + 1) * VROW + kg * 16;
    *(v8bf*)(v0)     = a0lo;
    *(v8bf*)(v0 + 8) = a0hi;
    *(v8bf*)(v1)     = a1lo;
    *(v8bf*)(v1 + 8) = a1hi;
  }
}

__global__ __launch_bounds__(128)
void sdpa_flash_bf16_wmma(const float* __restrict__ Q,
                          const float* __restrict__ K,
                          const float* __restrict__ V,
                          float* __restrict__ O) {
  __shared__ __bf16 Kt[2][32 * KROW];   // K tile, row-major [k][d]
  __shared__ __bf16 VtT[2][128 * VROW]; // V tile, transposed [d][k]
  __shared__ __bf16 Pt[4][16 * PROW];   // per-wave P transpose scratch

  const int tid  = threadIdx.x;
  const int lane = tid & 31;
  const int wv   = tid >> 5;            // wave 0..3
  const int nn   = lane & 15;           // column / row-in-tile index
  const int kb   = (lane < 16) ? 0 : 8; // K-group for A/B fragments
  const int ro   = (lane < 16) ? 0 : 8; // row offset for C/D fragments

  const int qb = blockIdx.x * BM;
  const size_t base = ((size_t)blockIdx.z * Hn + blockIdx.y) * (size_t)Sn * Dn;
  const float* Qb = Q + base;
  const float* Kb = K + base;
  const float* Vb = V + base;
  float*       Ob = O + base;

  const int myrow0 = qb + wv * 16;  // this wave's q rows: myrow0..myrow0+15

  // ---- Q fragments (A layout), scale = log2(e)/sqrt(D) folded in ----------
  const float scl = 1.4426950408889634f / 11.313708498984760f;
  v16bf qf[4];
  {
    const float* qr = Qb + (size_t)(myrow0 + nn) * Dn;
#pragma unroll
    for (int c = 0; c < 4; ++c) {
      v8fl f0 = *(const v8fl*)(qr + c * 32 + kb);
      v8fl f1 = *(const v8fl*)(qr + c * 32 + 16 + kb);
      v16bf r;
#pragma unroll
      for (int e = 0; e < 8; ++e) {
        r[e]     = (__bf16)(f0[e] * scl);
        r[e + 8] = (__bf16)(f1[e] * scl);
      }
      qf[c] = r;
    }
  }

  v8f acc[8];
#pragma unroll
  for (int c = 0; c < 8; ++c) acc[c] = zero8();
  float mrow[8], lrow[8];
#pragma unroll
  for (int v = 0; v < 8; ++v) { mrow[v] = -__builtin_inff(); lrow[v] = 0.0f; }

  const int nsteps = (qb + BM) / BN;  // causal: only k-tiles up to block end

  stage_tile(Kt[0], VtT[0], Kb, Vb, 0, tid);
  __syncthreads();

  for (int s = 0; s < nsteps; ++s) {
    if (s + 1 < nsteps)
      stage_tile(Kt[(s + 1) & 1], VtT[(s + 1) & 1], Kb, Vb, (s + 1) * BN, tid);

    const int kpos = s * BN;
    if (kpos <= myrow0 + 15) {  // wave-uniform: EXEC stays all-1s inside
      const __bf16* Kbuf = Kt[s & 1];
      const __bf16* Vbuf = VtT[s & 1];

      // ---- S = Q * K^T over D=128 (4 WMMAs per 16-col tile) --------------
      v8f s0 = zero8(), s1 = zero8();
#pragma unroll
      for (int c = 0; c < 4; ++c) {
        v16bf b0 = frag16(Kbuf + nn * KROW + c * 32, kb);
        v16bf b1 = frag16(Kbuf + (16 + nn) * KROW + c * 32, kb);
        s0 = wmma_bf16(qf[c], b0, s0);
        s1 = wmma_bf16(qf[c], b1, s1);
      }

      // ---- causal mask (diagonal steps only) -----------------------------
      if (kpos + BN - 1 > myrow0) {
#pragma unroll
        for (int v = 0; v < 8; ++v) {
          const int r = myrow0 + ro + v;
          if (kpos + nn > r)      s0[v] = -__builtin_inff();
          if (kpos + 16 + nn > r) s1[v] = -__builtin_inff();
        }
      }

      // ---- online softmax (log2 domain) ----------------------------------
      float corr[8], rs[8];
#pragma unroll
      for (int v = 0; v < 8; ++v) {
        float mx = fmaxf(s0[v], s1[v]);
#pragma unroll
        for (int off = 1; off < 16; off <<= 1)
          mx = fmaxf(mx, __shfl_xor(mx, off, 32));
        const float nm = fmaxf(mrow[v], mx);
        corr[v] = __builtin_amdgcn_exp2f(mrow[v] - nm);
        mrow[v] = nm;
        const float p0 = __builtin_amdgcn_exp2f(s0[v] - nm);
        const float p1 = __builtin_amdgcn_exp2f(s1[v] - nm);
        s0[v] = p0;
        s1[v] = p1;
        float r2 = p0 + p1;
#pragma unroll
        for (int off = 1; off < 16; off <<= 1)
          r2 += __shfl_xor(r2, off, 32);
        rs[v] = r2;
      }
#pragma unroll
      for (int v = 0; v < 8; ++v) lrow[v] = lrow[v] * corr[v] + rs[v];
#pragma unroll
      for (int c = 0; c < 8; ++c)
#pragma unroll
        for (int v = 0; v < 8; ++v) acc[c][v] *= corr[v];

      // ---- P: C layout -> LDS -> A layout --------------------------------
      __bf16* pb = &Pt[wv][0];
#pragma unroll
      for (int v = 0; v < 8; ++v) {
        pb[(ro + v) * PROW + nn]      = (__bf16)s0[v];
        pb[(ro + v) * PROW + 16 + nn] = (__bf16)s1[v];
      }
      v16bf pa = frag16(pb + nn * PROW, kb);

      // ---- O += P * V (8 WMMAs across D=128), V^T rows are contiguous ----
#pragma unroll
      for (int c = 0; c < 8; ++c) {
        v16bf vbf = frag16(Vbuf + (c * 16 + nn) * VROW, kb);
        acc[c] = wmma_bf16(pa, vbf, acc[c]);
      }
    }
    __syncthreads();
  }

  // ---- epilogue: normalize by row sum and store fp32 ----------------------
  float inv[8];
#pragma unroll
  for (int v = 0; v < 8; ++v) inv[v] = 1.0f / lrow[v];
#pragma unroll
  for (int c = 0; c < 8; ++c) {
#pragma unroll
    for (int v = 0; v < 8; ++v) {
      Ob[(size_t)(myrow0 + ro + v) * Dn + c * 16 + nn] = acc[c][v] * inv[v];
    }
  }
}

extern "C" void kernel_launch(void* const* d_in, const int* in_sizes, int n_in,
                              void* d_out, int out_size, void* d_ws, size_t ws_size,
                              hipStream_t stream) {
  (void)in_sizes; (void)n_in; (void)out_size; (void)d_ws; (void)ws_size;
  const float* Q = (const float*)d_in[0];
  const float* K = (const float*)d_in[1];
  const float* V = (const float*)d_in[2];
  // d_in[3] is the causal mask; causality is computed analytically in-kernel.
  float* O = (float*)d_out;

  dim3 grid(Sn / BM, Hn, Bn);
  dim3 block(128);
  hipLaunchKernelGGL(sdpa_flash_bf16_wmma, grid, block, 0, stream, Q, K, V, O);
}
